// LSTMDecoder_24292335027033
// MI455X (gfx1250) — matrix operand order repaired and checked by reading, not compiled, not run on hardware
//
#include <hip/hip_runtime.h>
#include <hip/hip_bf16.h>
#include <stdint.h>

// Problem dims (match reference)
#define B_    256
#define T_    64
#define DIN_  1024
#define H_    1024
#define E_    256
#define V_    8192
#define KCAT_ (DIN_ + E_ + H_)   // 2304 : [x_t | prev | h]
#define G4H_  (4 * H_)           // 4096
#define KC    64                 // K-chunk staged in LDS per iteration
#define AS3   __attribute__((address_space(3)))

typedef __attribute__((ext_vector_type(16))) __bf16 v16bf;
typedef __attribute__((ext_vector_type(8)))  float  v8f;

union Frag { v16bf v; uint4 q[2]; };

static __device__ __forceinline__ __bf16 f2bf(float f) {
  unsigned u = __builtin_bit_cast(unsigned, f);
  unsigned r = (u + 0x7FFFu + ((u >> 16) & 1u)) >> 16;  // round-to-nearest-even
  unsigned short s = (unsigned short)r;
  return __builtin_bit_cast(__bf16, s);
}
static __device__ __forceinline__ float sigf(float x) {
  return 1.0f / (1.0f + __expf(-x));
}

// Async stage 32 bytes (two b128, IOFFSET applies to both LDS and global side)
static __device__ __forceinline__ void async_copy32(void* lds_dst, const void* gsrc) {
  AS3 void* lp = (AS3 void*)lds_dst;
  asm volatile("global_load_async_to_lds_b128 %0, %1, off\n\t"
               "global_load_async_to_lds_b128 %0, %1, off offset:16"
               :: "v"(lp), "v"(gsrc) : "memory");
}
static __device__ __forceinline__ void wait_async0() {
  asm volatile("s_wait_asynccnt 0x0" ::: "memory");
}

// ---------------- one-time prep ----------------

__global__ void prep_weights(const float* __restrict__ W_ih, const float* __restrict__ W_hh,
                             const float* __restrict__ b_ih, const float* __restrict__ b_hh,
                             const float* __restrict__ lin_W,
                             __bf16* __restrict__ Wcat, __bf16* __restrict__ linWb,
                             float* __restrict__ bias) {
  int idx = blockIdx.x * blockDim.x + threadIdx.x;
  const int NW = G4H_ * KCAT_;
  if (idx < NW) {
    int n = idx / KCAT_, k = idx - n * KCAT_;
    float v = (k < DIN_ + E_) ? W_ih[(size_t)n * (DIN_ + E_) + k]
                              : W_hh[(size_t)n * H_ + (k - DIN_ - E_)];
    Wcat[idx] = f2bf(v);
  }
  if (idx < V_ * H_) linWb[idx] = f2bf(lin_W[idx]);
  if (idx < G4H_)    bias[idx]  = b_ih[idx] + b_hh[idx];
}

__global__ void prep_offsets(const int* __restrict__ sl, int* __restrict__ offs) {
  if (blockIdx.x == 0 && threadIdx.x == 0) {
    int acc = 0;
    for (int b = 0; b < B_; ++b) { offs[b] = acc; acc += sl[b]; }
  }
}

__global__ void init_state(const float* __restrict__ initT,
                           float* __restrict__ c, __bf16* __restrict__ h,
                           __bf16* __restrict__ prev) {
  int idx = blockIdx.x * blockDim.x + threadIdx.x;
  if (idx < B_ * H_) { c[idx] = 0.0f; h[idx] = f2bf(0.0f); }
  if (idx < B_ * E_) prev[idx] = f2bf(initT[idx & (E_ - 1)]);
}

// ---------------- per-step kernels ----------------

__global__ void pack_inp(const float* __restrict__ enc, const __bf16* __restrict__ prev,
                         const __bf16* __restrict__ h, __bf16* __restrict__ inp, int t) {
  int idx = blockIdx.x * blockDim.x + threadIdx.x;
  if (idx >= B_ * KCAT_) return;
  int b = idx / KCAT_, k = idx - b * KCAT_;
  __bf16 v;
  if (k < DIN_)           v = f2bf(enc[(size_t)b * T_ * DIN_ + (size_t)t * DIN_ + k]);
  else if (k < DIN_ + E_) v = prev[(size_t)b * E_ + (k - DIN_)];
  else                    v = h[(size_t)b * H_ + (k - DIN_ - E_)];
  inp[idx] = v;
}

// gates = inp @ Wcat^T + bias, fused LSTM cell.
// WG = 4 waves, 32 output rows. Each wave owns one 16-col N-tile and computes
// 2 M-tiles x 4 gates => each B fragment is reused by 2 WMMAs; A tile staged
// in LDS via async-load-to-LDS and shared by all 4 waves.
__global__ __launch_bounds__(128) void gemm_gates_lstm(
    const __bf16* __restrict__ inp, const __bf16* __restrict__ Wcat,
    const float* __restrict__ bias, float* __restrict__ c, __bf16* __restrict__ h) {
  __shared__ __align__(16) __bf16 sA[32][KC];
  const int lane = threadIdx.x & 31;
  const int wave = threadIdx.x >> 5;
  const int m0 = blockIdx.y * 32;
  const int n0 = (blockIdx.x * 4 + wave) * 16;
  const int col = lane & 15;   // B-frag: weight row / C-frag: N index
  const int hi  = lane >> 4;   // K-half selector: k = 16*hi + s (contiguous)

  v8f acc[4][2];
  #pragma unroll
  for (int g = 0; g < 4; ++g) {
    float bv = bias[g * H_ + n0 + col];
    #pragma unroll
    for (int mt = 0; mt < 2; ++mt)
      #pragma unroll
      for (int r = 0; r < 8; ++r) acc[g][mt][r] = bv;
  }
  const __bf16* wrow[4];
  #pragma unroll
  for (int g = 0; g < 4; ++g)
    wrow[g] = Wcat + (size_t)(g * H_ + n0 + col) * KCAT_ + hi * 16;

  const int ldrow = threadIdx.x >> 2;        // 32 rows
  const int ldseg = (threadIdx.x & 3) * 16;  // 32B (2 x uint4) per thread

  for (int kc = 0; kc < KCAT_; kc += KC) {
    __syncthreads();                         // previous chunk fully consumed
    async_copy32(&sA[ldrow][ldseg],
                 &inp[(size_t)(m0 + ldrow) * KCAT_ + kc + ldseg]);
    wait_async0();
    __syncthreads();                         // tile visible to all waves
    if (kc + KC < KCAT_) {
      #pragma unroll
      for (int g = 0; g < 4; ++g) __builtin_prefetch(wrow[g] + kc + KC, 0, 1);
    }
    #pragma unroll
    for (int ks = 0; ks < KC; ks += 32) {
      Frag a0, a1;                           // A: 16x32 bf16, k = 16*hi + s (contiguous)
      a0.q[0] = *(const uint4*)&sA[col][ks + hi * 16];
      a0.q[1] = *(const uint4*)&sA[col][ks + hi * 16 + 8];
      a1.q[0] = *(const uint4*)&sA[16 + col][ks + hi * 16];
      a1.q[1] = *(const uint4*)&sA[16 + col][ks + hi * 16 + 8];
      #pragma unroll
      for (int g = 0; g < 4; ++g) {
        Frag b;                              // B: lane = 16 contiguous bf16 of one W row
        const __bf16* p = wrow[g] + kc + ks;
        b.q[0] = *(const uint4*)p;
        b.q[1] = *(const uint4*)(p + 8);
        acc[g][0] = __builtin_amdgcn_wmma_f32_16x16x32_bf16(
            false, a0.v, false, b.v, (short)0, acc[g][0], false, false);
        acc[g][1] = __builtin_amdgcn_wmma_f32_16x16x32_bf16(
            false, a1.v, false, b.v, (short)0, acc[g][1], false, false);
      }
    }
  }
  // fused LSTM pointwise: i,f,g,o live in acc[0..3] at identical (m,n) slots
  #pragma unroll
  for (int mt = 0; mt < 2; ++mt) {
    #pragma unroll
    for (int r = 0; r < 8; ++r) {
      int m = m0 + mt * 16 + hi * 8 + r;
      int j = n0 + col;
      size_t off = (size_t)m * H_ + j;
      float iv = acc[0][mt][r], fv = acc[1][mt][r];
      float gv = acc[2][mt][r], ov = acc[3][mt][r];
      float cn = sigf(fv) * c[off] + sigf(iv) * tanhf(gv);
      c[off] = cn;
      h[off] = f2bf(sigf(ov) * tanhf(cn));
    }
  }
}

// logits = h @ lin_W^T + lin_b ; scatter rows to ragged d_out ; f32 scratch for argmax
__global__ __launch_bounds__(128) void gemm_logits(
    const __bf16* __restrict__ h, const __bf16* __restrict__ linW,
    const float* __restrict__ lin_b, const int* __restrict__ sl,
    const int* __restrict__ offs, float* __restrict__ logits,
    float* __restrict__ out, int t) {
  __shared__ __align__(16) __bf16 sA[32][KC];
  const int lane = threadIdx.x & 31;
  const int wave = threadIdx.x >> 5;
  const int m0 = blockIdx.y * 32;
  const int n0 = (blockIdx.x * 4 + wave) * 16;
  const int col = lane & 15, hi = lane >> 4;

  v8f acc[2];
  {
    float bv = lin_b[n0 + col];
    #pragma unroll
    for (int mt = 0; mt < 2; ++mt)
      #pragma unroll
      for (int r = 0; r < 8; ++r) acc[mt][r] = bv;
  }
  const __bf16* wrow = linW + (size_t)(n0 + col) * H_ + hi * 16;
  const int ldrow = threadIdx.x >> 2, ldseg = (threadIdx.x & 3) * 16;

  for (int kc = 0; kc < H_; kc += KC) {
    __syncthreads();
    async_copy32(&sA[ldrow][ldseg],
                 &h[(size_t)(m0 + ldrow) * H_ + kc + ldseg]);
    wait_async0();
    __syncthreads();
    if (kc + KC < H_) __builtin_prefetch(wrow + kc + KC, 0, 1);
    #pragma unroll
    for (int ks = 0; ks < KC; ks += 32) {
      Frag a0, a1, b;
      a0.q[0] = *(const uint4*)&sA[col][ks + hi * 16];
      a0.q[1] = *(const uint4*)&sA[col][ks + hi * 16 + 8];
      a1.q[0] = *(const uint4*)&sA[16 + col][ks + hi * 16];
      a1.q[1] = *(const uint4*)&sA[16 + col][ks + hi * 16 + 8];
      const __bf16* p = wrow + kc + ks;
      b.q[0] = *(const uint4*)p;
      b.q[1] = *(const uint4*)(p + 8);
      acc[0] = __builtin_amdgcn_wmma_f32_16x16x32_bf16(
          false, a0.v, false, b.v, (short)0, acc[0], false, false);
      acc[1] = __builtin_amdgcn_wmma_f32_16x16x32_bf16(
          false, a1.v, false, b.v, (short)0, acc[1], false, false);
    }
  }
  #pragma unroll
  for (int mt = 0; mt < 2; ++mt) {
    #pragma unroll
    for (int r = 0; r < 8; ++r) {
      int m = m0 + mt * 16 + hi * 8 + r;
      int n = n0 + col;
      float v = acc[mt][r];
      logits[(size_t)m * V_ + n] = v;
      if (t < sl[m]) out[((size_t)offs[m] + t) * V_ + n] = v;   // ragged scatter
    }
  }
}

// per-row argmax over V (first-index tie-break), then prev[b] = bf16(embedding[idx])
__global__ __launch_bounds__(256) void argmax_embed(
    const float* __restrict__ logits, const float* __restrict__ emb,
    __bf16* __restrict__ prev) {
  __shared__ float sm[256];
  __shared__ int   si[256];
  int b = blockIdx.x, tid = threadIdx.x;
  float best = -3.4e38f; int bi = 0x7FFFFFFF;
  for (int j = tid; j < V_; j += 256) {
    float v = logits[(size_t)b * V_ + j];
    if (v > best) { best = v; bi = j; }
  }
  sm[tid] = best; si[tid] = bi;
  __syncthreads();
  for (int s = 128; s > 0; s >>= 1) {
    if (tid < s) {
      if (sm[tid + s] > sm[tid] ||
          (sm[tid + s] == sm[tid] && si[tid + s] < si[tid])) {
        sm[tid] = sm[tid + s]; si[tid] = si[tid + s];
      }
    }
    __syncthreads();
  }
  int idx = si[0];
  prev[(size_t)b * E_ + tid] = f2bf(emb[(size_t)idx * E_ + tid]);
}

// ---------------- host ----------------

extern "C" void kernel_launch(void* const* d_in, const int* in_sizes, int n_in,
                              void* d_out, int out_size, void* d_ws, size_t ws_size,
                              hipStream_t stream) {
  const float* enc   = (const float*)d_in[0];
  const float* emb   = (const float*)d_in[1];
  const float* initT = (const float*)d_in[2];
  const float* W_ih  = (const float*)d_in[3];
  const float* W_hh  = (const float*)d_in[4];
  const float* b_ih  = (const float*)d_in[5];
  const float* b_hh  = (const float*)d_in[6];
  const float* lin_W = (const float*)d_in[7];
  const float* lin_b = (const float*)d_in[8];
  const int*   sl    = (const int*)d_in[9];
  float* out = (float*)d_out;

  char* p = (char*)d_ws;
  auto alloc = [&](size_t bytes) {
    char* r = p; p += (bytes + 255) & ~(size_t)255; return r;
  };
  __bf16* Wcat  = (__bf16*)alloc((size_t)G4H_ * KCAT_ * 2);  // 18.9 MB
  __bf16* linWb = (__bf16*)alloc((size_t)V_ * H_ * 2);       // 16.8 MB
  float*  bias  = (float*) alloc((size_t)G4H_ * 4);
  __bf16* inp   = (__bf16*)alloc((size_t)B_ * KCAT_ * 2);
  __bf16* hbuf  = (__bf16*)alloc((size_t)B_ * H_ * 2);
  float*  cbuf  = (float*) alloc((size_t)B_ * H_ * 4);
  float*  lg    = (float*) alloc((size_t)B_ * V_ * 4);       // 8 MB scratch logits
  __bf16* prev  = (__bf16*)alloc((size_t)B_ * E_ * 2);
  int*    offs  = (int*)   alloc((size_t)B_ * 4);

  hipLaunchKernelGGL(prep_weights, dim3((G4H_ * KCAT_ + 255) / 256), dim3(256), 0, stream,
                     W_ih, W_hh, b_ih, b_hh, lin_W, Wcat, linWb, bias);
  hipLaunchKernelGGL(prep_offsets, dim3(1), dim3(1), 0, stream, sl, offs);
  hipLaunchKernelGGL(init_state, dim3((B_ * H_ + 255) / 256), dim3(256), 0, stream,
                     initT, cbuf, hbuf, prev);

  for (int t = 0; t < T_; ++t) {   // recurrence serialized by stream order
    hipLaunchKernelGGL(pack_inp, dim3((B_ * KCAT_ + 255) / 256), dim3(256), 0, stream,
                       enc, prev, hbuf, inp, t);
    hipLaunchKernelGGL(gemm_gates_lstm, dim3(H_ / 64, B_ / 32), dim3(128), 0, stream,
                       inp, Wcat, bias, cbuf, hbuf);
    hipLaunchKernelGGL(gemm_logits, dim3(V_ / 64, B_ / 32), dim3(128), 0, stream,
                       hbuf, linWb, lin_b, sl, offs, lg, out, t);
    hipLaunchKernelGGL(argmax_embed, dim3(B_), dim3(256), 0, stream, lg, emb, prev);
  }
  (void)in_sizes; (void)n_in; (void)out_size; (void)ws_size;
}
